// FOGWithJK_53558242181882
// MI455X (gfx1250) — matrix operand order repaired and checked by reading, not compiled, not run on hardware
//
#include <hip/hip_runtime.h>

typedef __attribute__((ext_vector_type(16))) __bf16 v16bf;
typedef __attribute__((ext_vector_type(8)))  float  v8f;

#define LRELU_SLOPE 0.01f
#define BN_EPS 1e-5f

// ---------------------------------------------------------------------------
// helpers
// ---------------------------------------------------------------------------
__device__ __forceinline__ unsigned short f2bfbits(float x) {
  union { float f; unsigned int u; } v; v.f = x;
  unsigned int u = v.u;
  u += 0x7FFFu + ((u >> 16) & 1u);   // round-to-nearest-even
  return (unsigned short)(u >> 16);
}

__device__ __forceinline__ v16bf cvt16(float4 a, float4 b, float4 c, float4 d) {
  v16bf r;
  r[0]=(__bf16)(a.x);  r[1]=(__bf16)(a.y);  r[2]=(__bf16)(a.z);  r[3]=(__bf16)(a.w);
  r[4]=(__bf16)(b.x);  r[5]=(__bf16)(b.y);  r[6]=(__bf16)(b.z);  r[7]=(__bf16)(b.w);
  r[8]=(__bf16)(c.x);  r[9]=(__bf16)(c.y);  r[10]=(__bf16)(c.z); r[11]=(__bf16)(c.w);
  r[12]=(__bf16)(d.x); r[13]=(__bf16)(d.y); r[14]=(__bf16)(d.z); r[15]=(__bf16)(d.w);
  return r;
}
__device__ __forceinline__ v16bf scale_cvt16(float s, float4 a, float4 b, float4 c, float4 d) {
  v16bf r;
  r[0]=(__bf16)(s*a.x);  r[1]=(__bf16)(s*a.y);  r[2]=(__bf16)(s*a.z);  r[3]=(__bf16)(s*a.w);
  r[4]=(__bf16)(s*b.x);  r[5]=(__bf16)(s*b.y);  r[6]=(__bf16)(s*b.z);  r[7]=(__bf16)(s*b.w);
  r[8]=(__bf16)(s*c.x);  r[9]=(__bf16)(s*c.y);  r[10]=(__bf16)(s*c.z); r[11]=(__bf16)(s*c.w);
  r[12]=(__bf16)(s*d.x); r[13]=(__bf16)(s*d.y); r[14]=(__bf16)(s*d.z); r[15]=(__bf16)(s*d.w);
  return r;
}

// ---------------------------------------------------------------------------
// generic small kernels
// ---------------------------------------------------------------------------
__global__ void k_zero(float* __restrict__ p, int n) {
  int i = blockIdx.x * blockDim.x + threadIdx.x;
  if (i < n) p[i] = 0.0f;
}

// out[n,h] = leaky(sum_k in[n,k]*W[h,k] (+bias[h]));  slope==1.0 -> identity
__global__ void k_gemm_act(float* __restrict__ out, const float* __restrict__ in,
                           const float* __restrict__ W, const float* __restrict__ bias,
                           int n, int H, int K, float slope) {
  int idx = blockIdx.x * blockDim.x + threadIdx.x;
  if (idx >= n * H) return;
  int h = idx % H, row = idx / H;
  const float4* a = (const float4*)(in + (size_t)row * K);
  const float4* w = (const float4*)(W + (size_t)h * K);
  float s = 0.0f;
  for (int k = 0; k < (K >> 2); ++k) {
    float4 x = a[k], y = w[k];
    s += x.x*y.x + x.y*y.y + x.z*y.z + x.w*y.w;
  }
  if (bias) s += bias[h];
  out[idx] = (s > 0.0f) ? s : s * slope;
}

// per-column sum / sumsq accumulated into stats[0..C-1], stats[C..2C-1]
__global__ void k_colstats(const float* __restrict__ x, int n, int C,
                           float* __restrict__ stats) {
  __shared__ float ls[256], ls2[256];
  int tid = threadIdx.x;
  int col = tid % C, sub = tid / C;
  int rowsPerBlk = blockDim.x / C;
  float s = 0.0f, s2 = 0.0f;
  for (int r = blockIdx.x * rowsPerBlk + sub; r < n; r += gridDim.x * rowsPerBlk) {
    float v = x[(size_t)r * C + col];
    s += v; s2 += v * v;
  }
  ls[tid] = s; ls2[tid] = s2;
  __syncthreads();
  for (int st = blockDim.x >> 1; st >= C; st >>= 1) {
    if (tid < st) { ls[tid] += ls[tid + st]; ls2[tid] += ls2[tid + st]; }
    __syncthreads();
  }
  if (tid < C) {
    atomicAdd(&stats[col], ls[col]);
    atomicAdd(&stats[C + col], ls2[col]);
  }
}

// ss[c] = g*rsqrt(var+eps); ss[C+c] = b - mean*scale   (biased var, like jnp)
__global__ void k_bn_finalize(const float* __restrict__ stats,
                              const float* __restrict__ g, const float* __restrict__ b,
                              int C, float n, float* __restrict__ ss) {
  int c = threadIdx.x;
  if (c >= C) return;
  float mean = stats[c] / n;
  float var  = stats[C + c] / n - mean * mean;
  float sc   = g[c] * rsqrtf(var + BN_EPS);
  ss[c] = sc;
  ss[C + c] = b[c] - mean * sc;
}

__global__ void k_bn_apply(float* __restrict__ out, const float* __restrict__ in,
                           const float* __restrict__ ss, int total, int C, int relu) {
  int i = blockIdx.x * blockDim.x + threadIdx.x;
  if (i >= total) return;
  int c = i % C;
  float v = in[i] * ss[c] + ss[C + c];
  if (relu) v = fmaxf(v, 0.0f);
  out[i] = v;
}

// agg[dst[e]*64+c] += xsrc[src[e]*64+c]
__global__ void k_scatter(float* __restrict__ agg, const float* __restrict__ xsrc,
                          const int* __restrict__ src, const int* __restrict__ dst, int E) {
  int idx = blockIdx.x * blockDim.x + threadIdx.x;
  if (idx >= E * 64) return;
  int e = idx >> 6, c = idx & 63;
  atomicAdd(agg + (size_t)dst[e] * 64 + c, xsrc[(size_t)src[e] * 64 + c]);
}

__global__ void k_add3(float* __restrict__ out, const float* __restrict__ a,
                       const float* __restrict__ b, const float* __restrict__ c, int n) {
  int i = blockIdx.x * blockDim.x + threadIdx.x;
  if (i < n) out[i] = a[i] + b[i] + c[i];
}

__global__ void k_logsoftmax40(float* __restrict__ out, const float* __restrict__ logits, int n) {
  int i = blockIdx.x * blockDim.x + threadIdx.x;
  if (i >= n) return;
  const float* l = logits + (size_t)i * 40;
  float m = l[0];
  for (int c = 1; c < 40; ++c) m = fmaxf(m, l[c]);
  float s = 0.0f;
  for (int c = 0; c < 40; ++c) s += expf(l[c] - m);
  float ls = logf(s);
  float* o = out + (size_t)i * 40;
  for (int c = 0; c < 40; ++c) o[c] = l[c] - m - ls;
}

// ---------------------------------------------------------------------------
// weight pre-pack into B-fragment-native bf16 layout.
// packed[((dt*nk + ks)*32 + lane)*16 + e] = bf16(W[dt*16 + lane%16][ks*32 + koffB])
// koffB = (e & 7) + (e >= 8 ? 16 : 0) + (lane>=16 ? 8 : 0)
// ---------------------------------------------------------------------------
__global__ void k_prepack_b(unsigned short* __restrict__ packed,
                            const float* __restrict__ W, int Dout, int K) {
  int idx = blockIdx.x * blockDim.x + threadIdx.x;
  if (idx >= Dout * K) return;
  int e    = idx & 15;
  int lane = (idx >> 4) & 31;
  int t    = idx >> 9;
  int nk   = K >> 5;
  int ks   = t % nk;
  int dt   = t / nk;
  int koff = (e & 7) + ((e & 8) ? 16 : 0) + ((lane & 16) ? 8 : 0);
  float v = W[(size_t)(dt * 16 + (lane & 15)) * K + ks * 32 + koff];
  packed[idx] = f2bfbits(v);
}

// ---------------------------------------------------------------------------
// FOG projection: out[n,d] = sum_{i,j} x_des[n,i]*agg[n,j]*Wp[d, i*64+j]
// A-fragments synthesized on the fly (feat never materialized).
// Each wave: 16 rows x 128 cols (8 d-tiles -> 64 acc VGPRs, no spill).
// Block = 4 waves: 2 row-tiles x 2 d-halves. K=4096 (128 K-steps).
// ---------------------------------------------------------------------------
__global__ void __launch_bounds__(128, 1)
k_fog_wmma(float* __restrict__ out, const float* __restrict__ xdes,
           const float* __restrict__ agg,
           const unsigned short* __restrict__ packedWp, int n) {
  int lane  = threadIdx.x & 31;
  int wave  = threadIdx.x >> 5;
  int row0  = (blockIdx.x * 2 + (wave >> 1)) * 16;
  int dbase = (wave & 1) * 8;                 // d-tile half owned by this wave
  if (row0 >= n) return;                      // wave-uniform
  int rr = lane & 15;
  int hi = lane >> 4;
  int r  = row0 + rr;

  // two agg fragments (fp32), A koff pattern: base=hi*8, e<8 -> +e, e>=8 -> +16+(e-8)
  const float* arow = agg + (size_t)r * 64 + hi * 8;
  float4 g00 = *(const float4*)(arow +  0);
  float4 g01 = *(const float4*)(arow +  4);
  float4 g02 = *(const float4*)(arow + 16);
  float4 g03 = *(const float4*)(arow + 20);
  float4 g10 = *(const float4*)(arow + 32);
  float4 g11 = *(const float4*)(arow + 36);
  float4 g12 = *(const float4*)(arow + 48);
  float4 g13 = *(const float4*)(arow + 52);

  const float* xrow = xdes + (size_t)r * 64;
  const v16bf* BP = (const v16bf*)packedWp;

  v8f acc[8];
#pragma unroll
  for (int dt = 0; dt < 8; ++dt) acc[dt] = v8f{0,0,0,0,0,0,0,0};

  for (int k2 = 0; k2 < 64; ++k2) {          // k2 = i index; 2 K-steps per iter
    float xd = xrow[k2];
    v16bf af0 = scale_cvt16(xd, g00, g01, g02, g03);   // j in [0,32)
#pragma unroll
    for (int dt = 0; dt < 8; ++dt) {
      v16bf bf = BP[((dbase + dt) * 128 + 2 * k2) * 32 + lane];
      acc[dt] = __builtin_amdgcn_wmma_f32_16x16x32_bf16(false, af0, false, bf,
                                                        (short)0, acc[dt], false, false);
    }
    v16bf af1 = scale_cvt16(xd, g10, g11, g12, g13);   // j in [32,64)
#pragma unroll
    for (int dt = 0; dt < 8; ++dt) {
      v16bf bf = BP[((dbase + dt) * 128 + 2 * k2 + 1) * 32 + lane];
      acc[dt] = __builtin_amdgcn_wmma_f32_16x16x32_bf16(false, af1, false, bf,
                                                        (short)0, acc[dt], false, false);
    }
  }

#pragma unroll
  for (int dt = 0; dt < 8; ++dt) {
    int col = (dbase + dt) * 16 + rr;
#pragma unroll
    for (int q = 0; q < 8; ++q)
      out[(size_t)(row0 + q + hi * 8) * 256 + col] = acc[dt][q];
  }
}

// ---------------------------------------------------------------------------
// plain WMMA GEMM: out[n,0..255] = relu(A[n,:K] @ W^T + bias)
// Each wave: 16 rows x 128 cols (8 d-tiles).
// ---------------------------------------------------------------------------
__global__ void __launch_bounds__(128, 1)
k_gemm_wmma_bias_relu(float* __restrict__ out, const float* __restrict__ A,
                      const unsigned short* __restrict__ packedW,
                      const float* __restrict__ bias, int n, int K) {
  int lane  = threadIdx.x & 31;
  int wave  = threadIdx.x >> 5;
  int row0  = (blockIdx.x * 2 + (wave >> 1)) * 16;
  int dbase = (wave & 1) * 8;
  if (row0 >= n) return;
  int rr = lane & 15;
  int hi = lane >> 4;
  int r  = row0 + rr;
  int nk = K >> 5;
  const v16bf* BP = (const v16bf*)packedW;

  v8f acc[8];
#pragma unroll
  for (int dt = 0; dt < 8; ++dt) acc[dt] = v8f{0,0,0,0,0,0,0,0};

  for (int ks = 0; ks < nk; ++ks) {
    const float* ap = A + (size_t)r * K + ks * 32 + hi * 8;
    float4 u0 = *(const float4*)(ap +  0);
    float4 u1 = *(const float4*)(ap +  4);
    float4 u2 = *(const float4*)(ap + 16);
    float4 u3 = *(const float4*)(ap + 20);
    v16bf af = cvt16(u0, u1, u2, u3);
#pragma unroll
    for (int dt = 0; dt < 8; ++dt) {
      v16bf bf = BP[((dbase + dt) * nk + ks) * 32 + lane];
      acc[dt] = __builtin_amdgcn_wmma_f32_16x16x32_bf16(false, af, false, bf,
                                                        (short)0, acc[dt], false, false);
    }
  }

#pragma unroll
  for (int dt = 0; dt < 8; ++dt) {
    int col = (dbase + dt) * 16 + rr;
    float bv = bias[col];
#pragma unroll
    for (int q = 0; q < 8; ++q) {
      float v = acc[dt][q] + bv;
      out[(size_t)(row0 + q + hi * 8) * 256 + col] = fmaxf(v, 0.0f);
    }
  }
}

// ---------------------------------------------------------------------------
// host orchestration
// ---------------------------------------------------------------------------
static inline unsigned nblk(long long total, int b) { return (unsigned)((total + b - 1) / b); }

extern "C" void kernel_launch(void* const* d_in, const int* in_sizes, int n_in,
                              void* d_out, int out_size, void* d_ws, size_t ws_size,
                              hipStream_t stream) {
  (void)n_in; (void)out_size; (void)ws_size;
  const int N = in_sizes[0] / 128;
  const int E = in_sizes[1] / 2;

  const float* x      = (const float*)d_in[0];
  const int*   ei     = (const int*)d_in[1];
  const int*   srcIdx = ei;
  const int*   dstIdx = ei + E;
  const float* gbn[2] = {(const float*)d_in[23], (const float*)d_in[25]};
  const float* bbn[2] = {(const float*)d_in[24], (const float*)d_in[26]};
  const float* Wm1 = (const float*)d_in[27];
  const float* bm1 = (const float*)d_in[28];
  const float* Wm2 = (const float*)d_in[29];
  const float* bm2 = (const float*)d_in[30];

  // workspace carve-up (fp32 units)
  float* wsf = (float*)d_ws;
  size_t o = 0;
  float* xdes = wsf + o; o += (size_t)N * 64;
  float* xsrc = wsf + o; o += (size_t)N * 64;
  float* aggb = wsf + o; o += (size_t)N * 64;
  float* ybuf = wsf + o; o += (size_t)N * 64;
  float* hbuf = wsf + o; o += (size_t)N * 256;
  float* x1   = wsf + o; o += (size_t)N * 256;
  float* x2   = wsf + o; o += (size_t)N * 256;
  float* x3   = wsf + o; o += (size_t)N * 256;
  unsigned short* packed = (unsigned short*)(wsf + o); o += (size_t)(256 * 4096) / 2;
  float* stats = wsf + o; o += 512;
  float* ss    = wsf + o; o += 512;

  float* layer_out[3] = {hbuf, hbuf, x3};
  float* layer_relu[2] = {x1, x2};

  for (int l = 0; l < 3; ++l) {
    const float* W1 = (const float*)d_in[2 + 7 * l];
    const float* W2 = (const float*)d_in[3 + 7 * l];
    const float* Wp = (const float*)d_in[4 + 7 * l];
    const float* g1 = (const float*)d_in[5 + 7 * l];
    const float* b1 = (const float*)d_in[6 + 7 * l];
    const float* g2 = (const float*)d_in[7 + 7 * l];
    const float* b2 = (const float*)d_in[8 + 7 * l];
    const float* inX = (l == 0) ? x : ((l == 1) ? x1 : x2);
    const int Kin = (l == 0) ? 128 : 256;

    // x_des = BN(leaky(in @ W1^T))
    k_gemm_act<<<nblk((long long)N * 64, 256), 256, 0, stream>>>(ybuf, inX, W1, nullptr, N, 64, Kin, LRELU_SLOPE);
    k_zero<<<1, 128, 0, stream>>>(stats, 128);
    k_colstats<<<512, 256, 0, stream>>>(ybuf, N, 64, stats);
    k_bn_finalize<<<1, 64, 0, stream>>>(stats, g1, b1, 64, (float)N, ss);
    k_bn_apply<<<nblk((long long)N * 64, 256), 256, 0, stream>>>(xdes, ybuf, ss, N * 64, 64, 0);

    // x_src = BN(leaky(x_des @ W2^T))
    k_gemm_act<<<nblk((long long)N * 64, 256), 256, 0, stream>>>(ybuf, xdes, W2, nullptr, N, 64, 64, LRELU_SLOPE);
    k_zero<<<1, 128, 0, stream>>>(stats, 128);
    k_colstats<<<512, 256, 0, stream>>>(ybuf, N, 64, stats);
    k_bn_finalize<<<1, 64, 0, stream>>>(stats, g2, b2, 64, (float)N, ss);
    k_bn_apply<<<nblk((long long)N * 64, 256), 256, 0, stream>>>(xsrc, ybuf, ss, N * 64, 64, 0);

    // agg = segment_sum(x_src[src], dst)
    k_zero<<<nblk((long long)N * 64, 256), 256, 0, stream>>>(aggb, N * 64);
    k_scatter<<<nblk((long long)E * 64, 256), 256, 0, stream>>>(aggb, xsrc, srcIdx, dstIdx, E);

    // out = outer(x_des, agg).flat @ Wp^T   (WMMA, feat on the fly)
    k_prepack_b<<<nblk(256 * 4096, 256), 256, 0, stream>>>(packed, Wp, 256, 4096);
    k_fog_wmma<<<nblk(N, 32), 128, 0, stream>>>(layer_out[l], xdes, aggb, packed, N);

    if (l < 2) {
      k_zero<<<1, 512, 0, stream>>>(stats, 512);
      k_colstats<<<512, 256, 0, stream>>>(hbuf, N, 256, stats);
      k_bn_finalize<<<1, 256, 0, stream>>>(stats, gbn[l], bbn[l], 256, (float)N, ss);
      k_bn_apply<<<nblk((long long)N * 256, 256), 256, 0, stream>>>(layer_relu[l], hbuf, ss, N * 256, 256, 1);
    }
  }

  // head: s = x1+x2+x3; hid = relu(s@Wm1^T+bm1); logits = hid@Wm2^T+bm2; log_softmax
  k_add3<<<nblk((long long)N * 256, 256), 256, 0, stream>>>(hbuf, x1, x2, x3, N * 256);
  k_prepack_b<<<nblk(256 * 256, 256), 256, 0, stream>>>(packed, Wm1, 256, 256);
  k_gemm_wmma_bias_relu<<<nblk(N, 32), 128, 0, stream>>>(x1 /*hid*/, hbuf, packed, bm1, N, 256);
  k_gemm_act<<<nblk((long long)N * 40, 256), 256, 0, stream>>>(x2 /*logits*/, x1, Wm2, bm2, N, 40, 256, 1.0f);
  k_logsoftmax40<<<nblk(N, 256), 256, 0, stream>>>((float*)d_out, x2, N);
}
